// CFGEncoder_21449066676720
// MI455X (gfx1250) — compile-verified, matrix operand order checked by reading, not run on hardware
//
#include <hip/hip_runtime.h>
#include <hip/hip_bf16.h>

// ---------------- problem constants ----------------
#define BB 8
#define NN 1024
#define LL 16
#define DD 256
#define HH 4
#define DH 64
#define NL 2
#define NEG_BIG (-1.0e9f)

typedef __attribute__((ext_vector_type(16))) _Float16 v16h;
typedef __attribute__((ext_vector_type(8)))  float    v8f;

union Frag16 { v16h v; uint4 u[2]; _Float16 h[16]; };

#define WMMA_F16(A, B, C) \
  __builtin_amdgcn_wmma_f32_16x16x32_f16(false, (A), false, (B), (short)0, (C), false, false)

// ---- gfx1250 async global->LDS copy (ASYNCcnt path), guarded fallback ----
#if defined(__has_builtin)
#if __has_builtin(__builtin_amdgcn_global_load_async_to_lds_b128)
#define HAVE_ASYNC_LDS 1
#endif
#endif

typedef int v4i_ __attribute__((ext_vector_type(4)));
typedef __attribute__((address_space(3))) v4i_ lv4i;

__device__ __forceinline__ void copy16_async(const void* g, void* l) {
#ifdef HAVE_ASYNC_LDS
  __builtin_amdgcn_global_load_async_to_lds_b128(
      (v4i_*)g, (lv4i*)(uint32_t)(uintptr_t)l, 0, 0);
#else
  *(uint4*)l = *(const uint4*)g;
#endif
}

__device__ __forceinline__ void async_fence() {
#ifdef HAVE_ASYNC_LDS
#if __has_builtin(__builtin_amdgcn_s_wait_asynccnt)
  __builtin_amdgcn_s_wait_asynccnt(0);
#else
  asm volatile("s_wait_asynccnt 0" ::: "memory");
#endif
#endif
}

// ---------------------------------------------------------------------------
// 1) node features: mean over L token embeddings.  grid = B*N, block = 256
// ---------------------------------------------------------------------------
__global__ void embed_kernel(const int* __restrict__ nodes,
                             const float* __restrict__ emb,
                             float* __restrict__ h_out,
                             _Float16* __restrict__ h16_out) {
  const int row = blockIdx.x;            // b*N + n
  const int d   = threadIdx.x;           // 0..255
  __shared__ int toks[LL];
  if (threadIdx.x < LL) toks[threadIdx.x] = nodes[row * LL + threadIdx.x];
  __syncthreads();
  float s = 0.f;
#pragma unroll
  for (int t = 0; t < LL; ++t) s += emb[toks[t] * DD + d];
  s *= (1.0f / (float)LL);
  h_out[row * DD + d]   = s;
  h16_out[row * DD + d] = (_Float16)s;
}

// ---------------------------------------------------------------------------
// 2) convert one layer's W (H,D,DH) f32 -> column-major f16: Wct[c*D + d]
// ---------------------------------------------------------------------------
__global__ void wconv_kernel(const float* __restrict__ W,
                             _Float16* __restrict__ Wct) {
  const int idx = blockIdx.x * blockDim.x + threadIdx.x;   // over H*D*DH
  const int h = idx >> 14;
  const int d = (idx >> 6) & (DD - 1);
  const int k = idx & (DH - 1);
  Wct[(h * DH + k) * DD + d] = (_Float16)W[idx];
}

// ---------------------------------------------------------------------------
// 3) projection GEMM: Wh[b,h,n,k] = sum_d h[b,n,d] * W[h,d,k]
//    grid = (512, 4), block = 128 (4 waves).
// ---------------------------------------------------------------------------
__global__ void proj_kernel(const _Float16* __restrict__ hA,
                            const _Float16* __restrict__ Wct,
                            _Float16* __restrict__ Wh) {
  const int lane = threadIdx.x & 31;
  const int wave = threadIdx.x >> 5;
  const int l15  = lane & 15;
  const int hi   = lane >> 4;
  const int kb   = hi * 8;
  const int rowBase = blockIdx.x * 16;
  const int ct = blockIdx.y * 4 + wave;
  const int c0 = ct * 16;

  v8f acc = {};
#pragma unroll
  for (int kk = 0; kk < 8; ++kk) {
    const int kbase = kk * 32;
    Frag16 A, Bv;
    const uint4* ap = (const uint4*)(hA + (rowBase + l15) * DD + kbase + kb);
    A.u[0] = ap[0];
    A.u[1] = ap[2];
    const uint4* bp = (const uint4*)(Wct + (c0 + l15) * DD + kbase + kb);
    Bv.u[0] = bp[0];
    Bv.u[1] = bp[2];
    acc = WMMA_F16(A.v, Bv.v, acc);
  }
  const int b = rowBase >> 10;
  const int nbase = rowBase & (NN - 1);
  const int hh = c0 >> 6;
  const int k = (c0 & (DH - 1)) + l15;
#pragma unroll
  for (int r = 0; r < 8; ++r) {
    const int n = nbase + r + hi * 8;
    Wh[((b * HH + hh) * NN + n) * DH + k] = (_Float16)acc[r];
  }
}

// ---------------------------------------------------------------------------
// 4) src/dst attention logits per node. grid = B*H*N/256, block = 256
// ---------------------------------------------------------------------------
__global__ void srcdst_kernel(const _Float16* __restrict__ Wh,
                              const float* __restrict__ a_src,
                              const float* __restrict__ a_dst,
                              float* __restrict__ srcv,
                              float* __restrict__ dstv) {
  const int idx = blockIdx.x * blockDim.x + threadIdx.x;   // (b*H+h)*N + n
  const int h = (idx / NN) & (HH - 1);
  const _Float16* w = Wh + (size_t)idx * DH;
  float ss = 0.f, dd = 0.f;
#pragma unroll
  for (int k = 0; k < DH; ++k) {
    const float v = (float)w[k];
    ss += v * a_src[h * DH + k];
    dd += v * a_dst[h * DH + k];
  }
  srcv[idx] = ss;
  dstv[idx] = dd;
}

// ---------------------------------------------------------------------------
// 5) fused flash-style GAT attention.
//    grid = B*(N/16) = 512 blocks, block = 128 (4 waves).
//    Block owns one 16-row tile; wave w computes head w for those rows, so the
//    adjacency tile is staged ONCE per block and shared by all 4 heads
//    (adjacency = dominant HBM term; this cuts its traffic 4x vs per-head
//    blocks).  All staging uses global_load_async_to_lds_b128 (ASYNCcnt).
// ---------------------------------------------------------------------------
__global__ void attn_kernel(const float* __restrict__ srcv,
                            const float* __restrict__ dstv,
                            const _Float16* __restrict__ Wh,
                            const int* __restrict__ adj,
                            float* __restrict__ h_cat) {
  __shared__ _Float16 ldsWh[HH][32 * DH];  // row-major K chunk per head (16KB)
  __shared__ int      adjLds[16 * 32];     // 16 rows x 32 cols (2KB)
  __shared__ float    dstLds[HH][32];      // per-head dst chunk

  const int tid  = threadIdx.x;
  const int lane = tid & 31;
  const int wave = tid >> 5;
  const int l15  = lane & 15;
  const int hi   = lane >> 4;
  const int kb   = hi * 8;

  const int b      = blockIdx.x >> 6;          // /(N/16)
  const int n0base = (blockIdx.x & 63) * 16;   // block's row tile
  const int hh     = wave;                     // this wave's head
  const int rowg   = n0base + l15;             // node this lane scores

  const float srow = srcv[(b * HH + hh) * NN + rowg];
  const int*      adjg = adj  + ((size_t)(b * NN + n0base)) * NN;
  const _Float16* whb  = Wh   + (size_t)b * HH * NN * DH;
  const float*    dstb = dstv + (size_t)b * HH * NN;

  float m = -3.0e38f, s = 0.f;
  v8f accs[4] = {{}, {}, {}, {}};

  for (int step = 0; step < 32; ++step) {
    const int j0 = step * 32;

    __syncthreads();                 // previous chunk fully consumed
    {
      // adjacency tile: 16 rows x 32 ints = 128 b128 ops (one per thread)
      const int r = tid >> 3, qd = tid & 7;
      copy16_async(adjg + (size_t)r * NN + j0 + qd * 4, &adjLds[r * 32 + qd * 4]);
      // per-head dst chunks: 4 x 32 floats = 32 b128 ops
      if (tid < 32) {
        const int hd = tid >> 3, qq = tid & 7;
        copy16_async(dstb + (size_t)hd * NN + j0 + qq * 4, &dstLds[hd][qq * 4]);
      }
      // Wh chunks for all 4 heads: 4 x 32 x 64 f16 = 1024 b128 ops (8/thread)
#pragma unroll
      for (int i = 0; i < 8; ++i) {
        const int o  = tid * 8 + i;
        const int hd = o >> 8, rem = o & 255, k = rem >> 3, sg = rem & 7;
        copy16_async(whb + ((size_t)hd * NN + j0 + k) * DH + sg * 8,
                     &ldsWh[hd][k * DH + sg * 8]);
      }
      if (step < 31)
        __builtin_prefetch(whb + ((size_t)hh * NN + j0 + 32) * DH, 0, 0);
    }
    async_fence();                   // s_wait_asynccnt 0
    __syncthreads();

    // ---- masked leaky-relu scores (A-frag slot mapping) ----
    const int*   ap = &adjLds[l15 * 32];
    const float* dp = &dstLds[hh][0];
    float e[16];
#pragma unroll
    for (int q = 0; q < 16; ++q) {
      const int K = kb + q + ((q >= 8) ? 8 : 0);
      const int j = j0 + K;
      float ev = srow + dp[K];
      ev = (ev > 0.f) ? ev : 0.2f * ev;            // leaky_relu(0.2)
      if (ap[K] == 0 && j != rowg) ev = NEG_BIG;   // mask (self-loop on diag)
      e[q] = ev;
    }

    // ---- online softmax statistics (row split across lanes l and l^16) ----
    float tmax = e[0];
#pragma unroll
    for (int q = 1; q < 16; ++q) tmax = fmaxf(tmax, e[q]);
    tmax = fmaxf(tmax, __shfl_xor(tmax, 16));
    const float mnew  = fmaxf(m, tmax);
    const float scale = __expf(m - mnew);

    Frag16 P;
    float psum = 0.f;
#pragma unroll
    for (int q = 0; q < 16; ++q) {
      const float p = __expf(e[q] - mnew);
      psum += p;
      P.h[q] = (_Float16)p;
    }
    psum += __shfl_xor(psum, 16);
    s = s * scale + psum;
    m = mnew;

    // ---- rescale accumulators ----
    float sc[8];
#pragma unroll
    for (int r = 0; r < 8; ++r) sc[r] = __shfl(scale, r + hi * 8);
#pragma unroll
    for (int t = 0; t < 4; ++t)
#pragma unroll
      for (int r = 0; r < 8; ++r) accs[t][r] *= sc[r];

    // ---- p (16x32) @ Wh-chunk (32x64) via 4 WMMAs ----
    const _Float16* wl = &ldsWh[hh][0];
#pragma unroll
    for (int t = 0; t < 4; ++t) {
      Frag16 Bv;
      const int c = t * 16 + l15;
#pragma unroll
      for (int q = 0; q < 16; ++q) {
        const int K = kb + q + ((q >= 8) ? 8 : 0);
        Bv.h[q] = wl[K * DH + c];
      }
      accs[t] = WMMA_F16(P.v, Bv.v, accs[t]);
    }
  }

  // ---- finalize: divide by softmax denom, write h_cat (B,N,D) ------------
#pragma unroll
  for (int r = 0; r < 8; ++r) {
    const float inv = 1.0f / __shfl(s, r + hi * 8);
    const int n = n0base + r + hi * 8;
#pragma unroll
    for (int t = 0; t < 4; ++t) {
      h_cat[(size_t)(b * NN + n) * DD + hh * DH + t * 16 + l15] = accs[t][r] * inv;
    }
  }
}

// ---------------------------------------------------------------------------
// 6) residual + LayerNorm; also emits f16 copy for next layer's projection.
// ---------------------------------------------------------------------------
__global__ void ln_kernel(const float* __restrict__ h_in,
                          const float* __restrict__ h_cat,
                          const float* __restrict__ lnsc,
                          const float* __restrict__ lnbi,
                          float* __restrict__ h_out,
                          _Float16* __restrict__ h16_out) {
  const int row = blockIdx.x;
  const int d   = threadIdx.x;
  __shared__ float red[DD];
  const float v = h_in[row * DD + d] + h_cat[row * DD + d];
  red[d] = v;
  __syncthreads();
  for (int off = DD / 2; off > 0; off >>= 1) {
    if (d < off) red[d] += red[d + off];
    __syncthreads();
  }
  const float mu = red[0] * (1.0f / DD);
  __syncthreads();
  const float c = v - mu;
  red[d] = c * c;
  __syncthreads();
  for (int off = DD / 2; off > 0; off >>= 1) {
    if (d < off) red[d] += red[d + off];
    __syncthreads();
  }
  const float var = red[0] * (1.0f / DD);
  const float o = c * rsqrtf(var + 1e-5f) * lnsc[d] + lnbi[d];
  h_out[row * DD + d]   = o;
  h16_out[row * DD + d] = (_Float16)o;
}

// ---------------------------------------------------------------------------
// 7) output: h (f32) followed by cfg_len (int bits) in d_out.
// ---------------------------------------------------------------------------
__global__ void out_kernel(const float* __restrict__ h,
                           const int* __restrict__ len,
                           float* __restrict__ out, int nh) {
  const int i = blockIdx.x * blockDim.x + threadIdx.x;
  if (i < nh) {
    out[i] = h[i];
  } else if (i < nh + BB) {
    ((int*)out)[i] = len[i - nh];
  }
}

// ---------------------------------------------------------------------------
extern "C" void kernel_launch(void* const* d_in, const int* in_sizes, int n_in,
                              void* d_out, int out_size, void* d_ws, size_t ws_size,
                              hipStream_t stream) {
  const int*   cfg_adj   = (const int*)d_in[0];     // B*N*N
  const int*   cfg_nodes = (const int*)d_in[1];     // B*N*L
  const int*   cfg_len   = (const int*)d_in[2];     // B
  const float* emb       = (const float*)d_in[3];   // V*D
  const float* W         = (const float*)d_in[4];   // NL*H*D*DH
  const float* a_src     = (const float*)d_in[5];   // NL*H*DH
  const float* a_dst     = (const float*)d_in[6];   // NL*H*DH
  const float* ln_scale  = (const float*)d_in[7];   // NL*D
  const float* ln_bias   = (const float*)d_in[8];   // NL*D

  char* ws = (char*)d_ws;
  size_t off = 0;
  auto alloc = [&](size_t bytes) {
    char* p = ws + off;
    off += (bytes + 255) & ~(size_t)255;
    return p;
  };
  float*    node_h = (float*)   alloc((size_t)BB * NN * DD * 4);
  float*    h_cat  = (float*)   alloc((size_t)BB * NN * DD * 4);
  _Float16* h16    = (_Float16*)alloc((size_t)BB * NN * DD * 2);
  _Float16* Wh16   = (_Float16*)alloc((size_t)BB * HH * NN * DH * 2);
  _Float16* Wct    = (_Float16*)alloc((size_t)DD * DD * 2);
  float*    srcv   = (float*)   alloc((size_t)BB * HH * NN * 4);
  float*    dstv   = (float*)   alloc((size_t)BB * HH * NN * 4);

  embed_kernel<<<BB * NN, DD, 0, stream>>>(cfg_nodes, emb, node_h, h16);

  for (int l = 0; l < NL; ++l) {
    wconv_kernel<<<(HH * DD * DH) / 256, 256, 0, stream>>>(W + (size_t)l * HH * DD * DH, Wct);
    proj_kernel<<<dim3((BB * NN) / 16, 4), 128, 0, stream>>>(h16, Wct, Wh16);
    srcdst_kernel<<<(BB * HH * NN) / 256, 256, 0, stream>>>(
        Wh16, a_src + (size_t)l * HH * DH, a_dst + (size_t)l * HH * DH, srcv, dstv);
    attn_kernel<<<BB * (NN / 16), 128, 0, stream>>>(srcv, dstv, Wh16, cfg_adj, h_cat);
    ln_kernel<<<BB * NN, DD, 0, stream>>>(node_h, h_cat,
                                          ln_scale + (size_t)l * DD,
                                          ln_bias + (size_t)l * DD,
                                          node_h, h16);
  }

  const int nh = BB * NN * DD;
  out_kernel<<<(nh + BB + 255) / 256, 256, 0, stream>>>(node_h, cfg_len, (float*)d_out, nh);
}
